// AttentionLayer_13383118095164
// MI455X (gfx1250) — matrix operand order repaired and checked by reading, not compiled, not run on hardware
//
#include <hip/hip_runtime.h>
#include <hip/hip_bf16.h>

typedef __attribute__((ext_vector_type(16))) __bf16 v16bf;
typedef __attribute__((ext_vector_type(8)))  __bf16 v8bf;
typedef __attribute__((ext_vector_type(4)))  __bf16 v4bf;
typedef __attribute__((ext_vector_type(8)))  float  v8f;
typedef __attribute__((ext_vector_type(4)))  unsigned int v4u;
typedef __attribute__((ext_vector_type(8)))  int v8i;
typedef __attribute__((ext_vector_type(4)))  int v4i;

static __device__ __forceinline__ __bf16 f2bf(float f) { return (__bf16)f; }

// raw IEEE maximumNumber without the canonicalization pair fmaxf emits
static __device__ __forceinline__ float fmax_raw(float a, float b) {
    float r;
    asm("v_max_num_f32 %0, %1, %2" : "=v"(r) : "v"(a), "v"(b));
    return r;
}

static __device__ __forceinline__ v16bf combine8(v8bf lo, v8bf hi) {
    v16bf r;
#pragma unroll
    for (int i = 0; i < 8; ++i) { r[i] = lo[i]; r[8 + i] = hi[i]; }
    return r;
}

// ---------------------------------------------------------------------------
// Tensor Data Mover staging (CDNA5). Guarded so host pass / older toolchains
// fall back to manual copies.
// ---------------------------------------------------------------------------
#if defined(__HIP_DEVICE_COMPILE__) && __has_builtin(__builtin_amdgcn_tensor_load_to_lds) && __has_builtin(__builtin_amdgcn_s_wait_tensorcnt)
#define USE_TDM 1
#else
#define USE_TDM 0
#endif

#if USE_TDM
static __device__ __forceinline__ void tdm_issue(v4u g0, v8i g1) {
    v4i z4 = {};
#if defined(__clang_major__) && (__clang_major__ >= 23)
    v8i z8 = {};
    __builtin_amdgcn_tensor_load_to_lds(g0, g1, z4, z4, z8, 0);
#else
    __builtin_amdgcn_tensor_load_to_lds(g0, g1, z4, z4, 0);
#endif
}

// 2D tile: tile_d0 contiguous bf16 elems per row, tile_d1 rows, row stride
// `stride_elems` (bf16). LDS rows padded by TDM: pad_amount(+1) DWORDs after
// every 2^(pad_interval+1) DWORDs.
static __device__ __forceinline__ void tdm_load_2d(
    unsigned lds_addr, const __bf16* gptr, unsigned tile_d0, unsigned tile_d1,
    unsigned stride_elems, unsigned pad_interval, unsigned pad_amount)
{
    unsigned long long ga = (unsigned long long)(uintptr_t)gptr;
    v4u g0;
    g0.x = 1u;                                   // count=1 (valid user descriptor)
    g0.y = lds_addr;                             // lds_addr [63:32]
    g0.z = (unsigned)(ga & 0xffffffffu);         // global_addr [95:64]
    g0.w = (unsigned)((ga >> 32) & 0x01ffffffu)  // global_addr [120:96]
         | (2u << 30);                           // type=2 (image)
    v8i g1;
    g1[0] = (int)((1u << 16)                     // data_size = 2 bytes
                | (1u << 20)                     // pad_enable
                | (pad_interval << 22)
                | (pad_amount << 25));
    g1[1] = (int)((tile_d0 & 0xffffu) << 16);    // tensor_dim0[15:0] (== tile_d0)
    g1[2] = (int)(((tile_d0 >> 16) & 0xffffu) | ((tile_d1 & 0xffffu) << 16)); // td0 hi | tensor_dim1 lo
    g1[3] = (int)(((tile_d1 >> 16) & 0xffffu) | ((tile_d0 & 0xffffu) << 16)); // td1 hi | tile_dim0
    g1[4] = (int)(tile_d1 & 0xffffu);            // tile_dim1 | tile_dim2=0
    g1[5] = (int)stride_elems;                   // tensor_dim0_stride[31:0]
    g1[6] = 0;                                   // stride hi / dim1_stride lo
    g1[7] = 0;
    tdm_issue(g0, g1);
}
#endif

static __device__ __forceinline__ unsigned lds_off(const void* p) {
    // generic LDS pointer: addr[31:0] is the in-wave LDS byte offset
    return (unsigned)(uintptr_t)p;
}

// ---------------------------------------------------------------------------
// Elementwise f32 -> bf16 (activations). n must be a multiple of 4.
// ---------------------------------------------------------------------------
__global__ __launch_bounds__(256) void cvt_f32_bf16(
    const float* __restrict__ in, __bf16* __restrict__ out, long n)
{
    long i = ((long)blockIdx.x * 256 + threadIdx.x) * 4;
    if (i < n) {
        const float4 v = *(const float4*)(in + i);
        v4bf o;
        o[0] = f2bf(v.x); o[1] = f2bf(v.y); o[2] = f2bf(v.z); o[3] = f2bf(v.w);
        *(v4bf*)(out + i) = o;
    }
}

// ---------------------------------------------------------------------------
// Weight convert + transpose: W_f32[K,N] -> Wt_bf16[N,K]. K,N multiples of 64.
// ---------------------------------------------------------------------------
__global__ __launch_bounds__(256) void transpose_w(
    const float* __restrict__ W, __bf16* __restrict__ Wt, int K, int N)
{
    __shared__ __bf16 T[64][72];
    const int k0 = blockIdx.y * 64;
    const int n0 = blockIdx.x * 64;
    const int tid = threadIdx.x;
#pragma unroll
    for (int i = 0; i < 16; ++i) {
        int idx = tid + 256 * i;
        int kl = idx >> 6, nl = idx & 63;
        T[nl][kl] = f2bf(W[(size_t)(k0 + kl) * N + n0 + nl]);
    }
    __syncthreads();
#pragma unroll
    for (int i = 0; i < 16; ++i) {
        int idx = tid + 256 * i;
        int nl = idx >> 6, kl = idx & 63;
        Wt[(size_t)(n0 + nl) * K + k0 + kl] = T[nl][kl];
    }
}

// ---------------------------------------------------------------------------
// GEMM: C[M,N] = (A_bf16[M,K] * Wt_bf16[N,K]^T + bias) * oscale.
// 256 threads (8 waves), block tile 128Mx128N, K-step 32, double-buffered
// TDM staging into padded LDS rows (32 data + 16 pad bf16 = 96B, 32B aligned).
// Wave w: M strip (w>>2)*64 (4 M-tiles), N strip (w&3)*32 (2 N-tiles) -> 8 WMMA.
// ---------------------------------------------------------------------------
template <bool OUT_BF16>
__global__ __launch_bounds__(256) void gemm_tn(
    const __bf16* __restrict__ A, const __bf16* __restrict__ Wt,
    const float* __restrict__ bias, void* __restrict__ Cout,
    int M, int N, int Kd, float oscale)
{
    __shared__ __attribute__((aligned(32))) __bf16 As[2][128][48];
    __shared__ __attribute__((aligned(32))) __bf16 Bs[2][128][48];

    const int tid = threadIdx.x;
    const int lane = tid & 31;
    const int w = tid >> 5;
    const int lane15 = lane & 15;
    const int hi = lane >> 4;
    const int wm = (w >> 2) * 64;
    const int wn = (w & 3) * 32;
    const int m0 = blockIdx.y * 128;
    const int n0 = blockIdx.x * 128;
    const int kb8 = hi ? 8 : 0, kb16 = hi ? 16 : 0;
    const int nk = Kd >> 5;

    v8f acc[4][2] = {};

#if !USE_TDM
    auto stage = [&](int buf, int kk) {
#pragma unroll
        for (int i = 0; i < 2; ++i) {
            int c = tid + 256 * i;
            int row = c >> 2, kc = (c & 3) << 3;
            *(v8bf*)&As[buf][row][kc] = *(const v8bf*)(A + (size_t)(m0 + row) * Kd + kk + kc);
        }
#pragma unroll
        for (int i = 0; i < 2; ++i) {
            int c = tid + 256 * i;
            int row = c >> 2, kc = (c & 3) << 3;
            *(v8bf*)&Bs[buf][row][kc] = *(const v8bf*)(Wt + (size_t)(n0 + row) * Kd + kk + kc);
        }
    };
#endif

    // prologue: stage buffer 0
#if USE_TDM
    if (tid < 32) {
        tdm_load_2d(lds_off(&As[0][0][0]), A  + (size_t)m0 * Kd, 32u, 128u, (unsigned)Kd, 3u, 7u);
        tdm_load_2d(lds_off(&Bs[0][0][0]), Wt + (size_t)n0 * Kd, 32u, 128u, (unsigned)Kd, 3u, 7u);
    }
#else
    stage(0, 0);
#endif

    for (int kt = 0; kt < nk; ++kt) {
        const int cur = kt & 1;
#if USE_TDM
        if (tid < 32) __builtin_amdgcn_s_wait_tensorcnt(0);
#endif
        __syncthreads();                       // buffer `cur` ready for all waves
        if (kt + 1 < nk) {                     // overlap next DMA with compute
            const int kk = (kt + 1) << 5;
#if USE_TDM
            if (tid < 32) {
                tdm_load_2d(lds_off(&As[cur ^ 1][0][0]), A  + (size_t)m0 * Kd + kk, 32u, 128u, (unsigned)Kd, 3u, 7u);
                tdm_load_2d(lds_off(&Bs[cur ^ 1][0][0]), Wt + (size_t)n0 * Kd + kk, 32u, 128u, (unsigned)Kd, 3u, 7u);
            }
#else
            stage(cur ^ 1, kk);
#endif
        }

        v16bf bfr[2], afr[4];
#pragma unroll
        for (int nt = 0; nt < 2; ++nt)
            bfr[nt] = combine8(*(const v8bf*)&Bs[cur][wn + nt * 16 + lane15][kb8],
                               *(const v8bf*)&Bs[cur][wn + nt * 16 + lane15][16 + kb8]);
#pragma unroll
        for (int mt = 0; mt < 4; ++mt) {
            const int row = wm + mt * 16 + lane15;
            afr[mt] = combine8(*(const v8bf*)&As[cur][row][kb8],
                               *(const v8bf*)&As[cur][row][16 + kb8]);
        }
#pragma unroll
        for (int mt = 0; mt < 4; ++mt)
#pragma unroll
            for (int nt = 0; nt < 2; ++nt)
                acc[mt][nt] = __builtin_amdgcn_wmma_f32_16x16x32_bf16(
                    false, afr[mt], false, bfr[nt], (short)0, acc[mt][nt], false, false);
        __syncthreads();                       // done reading `cur` before its reuse
    }

#pragma unroll
    for (int nt = 0; nt < 2; ++nt) {
        const int gc = n0 + wn + nt * 16 + lane15;
        const float bv = bias[gc];
#pragma unroll
        for (int mt = 0; mt < 4; ++mt)
#pragma unroll
            for (int r = 0; r < 8; ++r) {
                const int gr = m0 + wm + mt * 16 + r + hi * 8;
                const float v = (acc[mt][nt][r] + bv) * oscale;
                if (OUT_BF16)
                    ((__bf16*)Cout)[(size_t)gr * N + gc] = f2bf(v);
                else
                    ((float*)Cout)[(size_t)gr * N + gc] = v;
            }
    }
    (void)M;
}

// ---------------------------------------------------------------------------
// Flash attention: block = (b, h, 64 L-rows), 4 waves x 16 rows.
// Q is pre-scaled by (1/sqrt(E))*log2(e) so softmax runs in base 2 with raw
// exp2f (v_exp_f32 is natively base-2). Row sums of P are computed on the
// matrix core (P x ones WMMA) instead of a shuffle tree; only the row max
// uses shuffles. K-tile via TDM (pad 64->80 elems), V-tile transposed
// manually, both double-buffered. Q,K,V,O bf16 [B,L/S,H*E].
// ---------------------------------------------------------------------------
__global__ __launch_bounds__(128) void attn_kernel(
    const __bf16* __restrict__ Q, const __bf16* __restrict__ Km,
    const __bf16* __restrict__ Vm, __bf16* __restrict__ O)
{
    constexpr int L = 2048, S = 1024, HE = 1024, E = 64;

    __shared__ __attribute__((aligned(32))) __bf16 Kt[2][32][80];     // [s][e] (pad 64->80)
    __shared__ __attribute__((aligned(32))) __bf16 Vt[2][64][48];     // [e][s] (pad 32->48)
    __shared__ __attribute__((aligned(32))) __bf16 Pb[4][16][48];     // per-wave P scratch

    const int tid = threadIdx.x, lane = tid & 31, w = tid >> 5;
    const int lane15 = lane & 15, hi = lane >> 4;
    const int b = blockIdx.z, h = blockIdx.y;
    const int l0 = blockIdx.x * 64 + w * 16;
    const int kb8 = hi ? 8 : 0, kb16 = hi ? 16 : 0;

    // Q fragments (e in [0,32) and [32,64))
    const size_t qbase = ((size_t)(b * L + l0 + lane15)) * HE + h * E;
    v16bf qf[2];
#pragma unroll
    for (int c = 0; c < 2; ++c) {
        int ee = c * 32;
        qf[c] = combine8(*(const v8bf*)(Q + qbase + ee + kb8),
                         *(const v8bf*)(Q + qbase + ee + 16 + kb8));
    }

    // all-ones B fragment: wmma(P, ones) broadcasts row sums of P into every
    // column of the result (same C-layout as lrun/acco -> no shuffles needed)
    v16bf onesb;
#pragma unroll
    for (int i = 0; i < 16; ++i) onesb[i] = (__bf16)1.0f;

    auto stage_v = [&](int buf, int s0) {       // batched loads, then stores
        v8bf vv[2];
#pragma unroll
        for (int i = 0; i < 2; ++i) {
            int c = tid + 128 * i;
            int si = c >> 3, ec = (c & 7) << 3;
            vv[i] = *(const v8bf*)(Vm + ((size_t)(b * S + s0 + si)) * HE + h * E + ec);
        }
#pragma unroll
        for (int i = 0; i < 2; ++i) {
            int c = tid + 128 * i;
            int si = c >> 3, ec = (c & 7) << 3;
#pragma unroll
            for (int j = 0; j < 8; ++j) Vt[buf][ec + j][si] = vv[i][j];
        }
    };
#if !USE_TDM
    auto stage_k = [&](int buf, int s0) {
        v8bf kv[2];
#pragma unroll
        for (int i = 0; i < 2; ++i) {
            int c = tid + 128 * i;
            int si = c >> 3, ec = (c & 7) << 3;
            kv[i] = *(const v8bf*)(Km + ((size_t)(b * S + s0 + si)) * HE + h * E + ec);
        }
#pragma unroll
        for (int i = 0; i < 2; ++i) {
            int c = tid + 128 * i;
            int si = c >> 3, ec = (c & 7) << 3;
            *(v8bf*)&Kt[buf][si][ec] = kv[i];
        }
    };
#endif

    float mrun[8], lrun[8];
#pragma unroll
    for (int r = 0; r < 8; ++r) { mrun[r] = -1e30f; lrun[r] = 0.f; }
    v8f acco[4] = {};

    // prologue: stage buffer 0 (s0 = 0)
#if USE_TDM
    if (tid < 32)
        tdm_load_2d(lds_off(&Kt[0][0][0]), Km + ((size_t)(b * S)) * HE + h * E,
                    64u, 32u, (unsigned)HE, 4u, 7u);
#else
    stage_k(0, 0);
#endif
    stage_v(0, 0);

    for (int s0 = 0, it = 0; s0 < S; s0 += 32, ++it) {
        const int cur = it & 1;
#if USE_TDM
        if (tid < 32) __builtin_amdgcn_s_wait_tensorcnt(0);
#endif
        __syncthreads();
        if (s0 + 32 < S) {
#if USE_TDM
            if (tid < 32)
                tdm_load_2d(lds_off(&Kt[cur ^ 1][0][0]),
                            Km + ((size_t)(b * S + s0 + 32)) * HE + h * E,
                            64u, 32u, (unsigned)HE, 4u, 7u);
#else
            stage_k(cur ^ 1, s0 + 32);
#endif
            stage_v(cur ^ 1, s0 + 32);
        }

        // scores (already scaled into base-2 domain via Q pre-scaling):
        // two 16x16 column tiles (s 0-15, 16-31), reduce over e=64
        v8f accs[2] = {};
#pragma unroll
        for (int c = 0; c < 2; ++c) {
            int ee = c * 32;
#pragma unroll
            for (int j = 0; j < 2; ++j) {
                v16bf kf = combine8(*(const v8bf*)&Kt[cur][j * 16 + lane15][ee + kb8],
                                    *(const v8bf*)&Kt[cur][j * 16 + lane15][ee + 16 + kb8]);
                accs[j] = __builtin_amdgcn_wmma_f32_16x16x32_bf16(
                    false, qf[c], false, kf, (short)0, accs[j], false, false);
            }
        }

        // online softmax in base 2; row max via shuffles in 16-lane halves
        float alv[8];
#pragma unroll
        for (int r = 0; r < 8; ++r) {
            float v0 = accs[0][r];
            float v1 = accs[1][r];
            float mx = fmax_raw(v0, v1);
#pragma unroll
            for (int msk = 1; msk <= 8; msk <<= 1) mx = fmax_raw(mx, __shfl_xor(mx, msk, 32));
            float nm = fmax_raw(mrun[r], mx);
            float al = exp2f(mrun[r] - nm);
            float p0 = exp2f(v0 - nm);
            float p1 = exp2f(v1 - nm);
            alv[r] = al;
            mrun[r] = nm;
#pragma unroll
            for (int nt = 0; nt < 4; ++nt) acco[nt][r] *= al;
            int Ml = r + hi * 8;
            Pb[w][Ml][lane15] = f2bf(p0);
            Pb[w][Ml][16 + lane15] = f2bf(p1);
        }

        // repack P (C-layout) into A-fragment layout via per-wave LDS scratch
        v16bf pf = combine8(*(const v8bf*)&Pb[w][lane15][kb8],
                            *(const v8bf*)&Pb[w][lane15][16 + kb8]);

        // row sums of P on the matrix core: srow[r] = sum_s P[row r][s]
        v8f zacc = {};
        v8f srow = __builtin_amdgcn_wmma_f32_16x16x32_bf16(
            false, pf, false, onesb, (short)0, zacc, false, false);
#pragma unroll
        for (int r = 0; r < 8; ++r) lrun[r] = lrun[r] * alv[r] + srow[r];

        // O += P[16x32] * V[32x64]
#pragma unroll
        for (int nt = 0; nt < 4; ++nt) {
            v16bf vf = combine8(*(const v8bf*)&Vt[cur][nt * 16 + lane15][kb8],
                                *(const v8bf*)&Vt[cur][nt * 16 + lane15][16 + kb8]);
            acco[nt] = __builtin_amdgcn_wmma_f32_16x16x32_bf16(
                false, pf, false, vf, (short)0, acco[nt], false, false);
        }
        __syncthreads();
    }

    // normalize and write O (bf16); one reciprocal per row
#pragma unroll
    for (int r = 0; r < 8; ++r) {
        const float rl = 1.0f / lrun[r];
        const int Ml = r + hi * 8;
        const size_t rowoff = ((size_t)(b * L + l0 + Ml)) * HE + h * E + lane15;
#pragma unroll
        for (int nt = 0; nt < 4; ++nt)
            O[rowoff + nt * 16] = f2bf(acco[nt][r] * rl);
    }
}

// ---------------------------------------------------------------------------
extern "C" void kernel_launch(void* const* d_in, const int* in_sizes, int n_in,
                              void* d_out, int out_size, void* d_ws, size_t ws_size,
                              hipStream_t stream) {
    const float* tgt = (const float*)d_in[0];   // [4,2048,1024]
    const float* src = (const float*)d_in[1];   // [4,1024,4096]
    const float* val = (const float*)d_in[2];   // [4,1024,4096]
    const float* Wq  = (const float*)d_in[3];
    const float* bq  = (const float*)d_in[4];
    const float* Wk  = (const float*)d_in[5];
    const float* bk  = (const float*)d_in[6];
    const float* Wv  = (const float*)d_in[7];
    const float* bv  = (const float*)d_in[8];
    const float* Wo  = (const float*)d_in[9];
    const float* bo  = (const float*)d_in[10];
    float* out = (float*)d_out;                 // [4,2048,1024] f32

    const size_t MB = 1u << 20;
    char* ws = (char*)d_ws;
    __bf16* Qw   = (__bf16*)(ws);               // 8192x1024 = 16 MB
    __bf16* Kw   = (__bf16*)(ws +  16 * MB);    // 4096x1024 =  8 MB
    __bf16* Vw   = (__bf16*)(ws +  24 * MB);    // 4096x1024 =  8 MB
    __bf16* Ow   = (__bf16*)(ws +  32 * MB);    // 8192x1024 = 16 MB
    __bf16* tgtb = (__bf16*)(ws +  48 * MB);    // 16 MB
    __bf16* srcb = (__bf16*)(ws +  64 * MB);    // 32 MB
    __bf16* valb = (__bf16*)(ws +  96 * MB);    // 32 MB
    __bf16* Wqt  = (__bf16*)(ws + 128 * MB);    // [1024,1024]  2 MB
    __bf16* Wkt  = (__bf16*)(ws + 130 * MB);    // [1024,4096]  8 MB
    __bf16* Wvt  = (__bf16*)(ws + 138 * MB);    // [1024,4096]  8 MB
    __bf16* Wot  = (__bf16*)(ws + 146 * MB);    // [1024,1024]  2 MB

    // softmax scale folded into Q: (1/sqrt(64)) * log2(e)
    const float QSCALE = 0.125f * 1.44269504088896f;

    // --- precision prep: one-time f32 -> bf16 (+ weight transpose) ---
    cvt_f32_bf16<<<8192, 256, 0, stream>>>(tgt, tgtb, 8388608L);
    cvt_f32_bf16<<<16384, 256, 0, stream>>>(src, srcb, 16777216L);
    cvt_f32_bf16<<<16384, 256, 0, stream>>>(val, valb, 16777216L);
    transpose_w<<<dim3(16, 16), 256, 0, stream>>>(Wq, Wqt, 1024, 1024);
    transpose_w<<<dim3(16, 64), 256, 0, stream>>>(Wk, Wkt, 4096, 1024);
    transpose_w<<<dim3(16, 64), 256, 0, stream>>>(Wv, Wvt, 4096, 1024);
    transpose_w<<<dim3(16, 16), 256, 0, stream>>>(Wo, Wot, 1024, 1024);

    // --- projections (bf16 WMMA, TDM-staged) ---
    gemm_tn<true><<<dim3(8, 64), 256, 0, stream>>>(tgtb, Wqt, bq, Qw, 8192, 1024, 1024, QSCALE);
    gemm_tn<true><<<dim3(8, 32), 256, 0, stream>>>(srcb, Wkt, bk, Kw, 4096, 1024, 4096, 1.0f);
    gemm_tn<true><<<dim3(8, 32), 256, 0, stream>>>(valb, Wvt, bv, Vw, 4096, 1024, 4096, 1.0f);

    // --- attention ---
    attn_kernel<<<dim3(32, 16, 4), 128, 0, stream>>>(Qw, Kw, Vw, Ow);

    // --- output projection (f32 out) ---
    gemm_tn<false><<<dim3(8, 64), 256, 0, stream>>>(Ow, Wot, bo, out, 8192, 1024, 1024, 1.0f);

    (void)in_sizes; (void)n_in; (void)out_size; (void)ws_size;
}
